// SparseFuseAttention_16862041604677
// MI455X (gfx1250) — matrix operand order repaired and checked by reading, not compiled, not run on hardware
//
#include <hip/hip_runtime.h>
#include <hip/hip_bf16.h>
#include <cstdint>

// ---------------------------------------------------------------------------
// MI455X (gfx1250) sparse fused attention.
// All matmuls run on v_wmma_f32_16x16x32_bf16 (wave32 WMMA), softmax in fp32.
// Non-transposing LDS staging uses global_load_async_to_lds_b128 (ASYNCcnt).
// ---------------------------------------------------------------------------

#define DIM    768
#define HEADS  12
#define HD     64
#define BATCH  8
#define SEQ    2121
#define NPAD   2128                 // 133 * 16
#define MROWS  (BATCH * NPAD)       // 17024 = 133 * 128
#define QKV_N  (3 * DIM)            // 2304
#define QSCALE 0.125f               // 64^-0.5
#define NTILES 135                  // query tiles per (b,h): 1 + 100 + 25 + 7 + 2

typedef __attribute__((ext_vector_type(16))) __bf16       v16bf;
typedef __attribute__((ext_vector_type(8)))  float        v8f;
typedef __attribute__((ext_vector_type(4)))  unsigned int u32x4;

union FragB {
    v16bf          v;
    unsigned short u[16];
    u32x4          q[2];
};

// Async DMA: LDS[ldsoff + imm] = MEM[sbase + voff + imm], 16B per lane.
#define ASYNC_LDS_B128(ldsoff, voff, sbase, imm)                               \
    asm volatile("global_load_async_to_lds_b128 %0, %1, %2 offset:" #imm       \
                 :: "v"(ldsoff), "v"(voff), "s"(sbase) : "memory")
#define WAIT_ASYNC0() asm volatile("s_wait_asynccnt 0x0" ::: "memory")

__device__ __forceinline__ unsigned short f2bf(float f) {
    union { float f; unsigned int u; } x;
    x.f = f;
    unsigned int r = x.u + 0x7FFFu + ((x.u >> 16) & 1u);   // round-to-nearest-even
    return (unsigned short)(r >> 16);
}

__device__ __forceinline__ v8f wmma_bf16(v16bf a, v16bf b, v8f c) {
    // D = A(16x32 bf16) * B(32x16 bf16) + C(16x16 f32)
    return __builtin_amdgcn_wmma_f32_16x16x32_bf16(false, a, false, b,
                                                   (short)0, c, false, false);
}

// ---------------------------------------------------------------------------
// fp32 -> bf16 conversions
// ---------------------------------------------------------------------------
__global__ void cvt_src_kernel(const float* __restrict__ src,
                               unsigned short* __restrict__ dst) {
    int i = blockIdx.x * 256 + threadIdx.x;
    const int total = BATCH * NPAD * DIM;
    if (i >= total) return;
    int d   = i % DIM;
    int row = i / DIM;
    int n   = row % NPAD;
    int b   = row / NPAD;
    float v = 0.0f;
    if (n < SEQ) v = src[((long)b * SEQ + n) * DIM + d];
    dst[i] = f2bf(v);
}

__global__ void cvt_w_kernel(const float* __restrict__ w,
                             unsigned short* __restrict__ dst, int count) {
    int i = blockIdx.x * 256 + threadIdx.x;
    if (i < count) dst[i] = f2bf(w[i]);
}

// ---------------------------------------------------------------------------
// bf16 GEMM: C[128x128 tile] = A[M x 768] * B[768 x NN]
//   mode 0: scatter into q/k/v [b][h][n][64] (scale q by QSCALE)
//   mode 1: out[b][n][c] = C + bias[c]   (fp32, only n < SEQ)
// 8 waves/WG, each wave owns a 32x64 sub-tile (2x4 WMMA accums).
// A tile is staged with async-to-LDS DMA (layout matches, no transpose);
// B tile needs a transpose so it goes through VGPRs.
// ---------------------------------------------------------------------------
__global__ __launch_bounds__(256) void gemm_bf16_kernel(
    const unsigned short* __restrict__ A,
    const unsigned short* __restrict__ Bm,
    int NN, int mode,
    unsigned short* __restrict__ qb,
    unsigned short* __restrict__ kb,
    unsigned short* __restrict__ vb,
    const float* __restrict__ bias,
    float* __restrict__ out)
{
    __shared__ unsigned short As[128 * 40];   // [m][k], k-contig, 80B row stride
    __shared__ unsigned short Bs[128 * 40];   // [n][k] (transposed), k-contig

    const int t    = threadIdx.x;
    const int lane = t & 31;
    const int w    = t >> 5;
    const int wm   = w & 3;        // 4 m-waves
    const int wn   = w >> 2;       // 2 n-waves
    const int tileM = blockIdx.x * 128;
    const int tileN = blockIdx.y * 128;
    const int mA = lane & 15, kh = lane >> 4;
    const unsigned long long Abase = (unsigned long long)(uintptr_t)A;

    v8f acc[2][4];
#pragma unroll
    for (int i = 0; i < 2; i++)
#pragma unroll
        for (int j = 0; j < 4; j++) acc[i][j] = {};

    for (int ks = 0; ks < DIM; ks += 32) {
        __syncthreads();
        {   // stage A tile 128x32 via async DMA (2 x 16B per thread)
            int m = t >> 1, kp = (t & 1) * 16;
            unsigned lds0 = (unsigned)(uintptr_t)&As[m * 40 + kp];
            unsigned goff = (unsigned)(((tileM + m) * DIM + ks + kp) * 2);
            ASYNC_LDS_B128(lds0, goff, Abase, 0);
            ASYNC_LDS_B128(lds0, goff, Abase, 16);
        }
        // stage B tile 32x128 transposed into [n][k]
#pragma unroll
        for (int j = 0; j < 16; j++) {
            int e = t + 256 * j;
            int kk = e >> 7, n = e & 127;
            Bs[n * 40 + kk] = Bm[(long)(ks + kk) * NN + tileN + n];
        }
        WAIT_ASYNC0();
        __syncthreads();

        FragB af[2], bf[4];
#pragma unroll
        for (int i = 0; i < 2; i++) {   // A frag: row=mA, j0-7 -> K=kh*8+j, j8-15 -> 16+kh*8+j
            int row = wm * 32 + i * 16 + mA;
            af[i].q[0] = *(const u32x4*)&As[row * 40 + kh * 8];
            af[i].q[1] = *(const u32x4*)&As[row * 40 + 16 + kh * 8];
        }
#pragma unroll
        for (int j = 0; j < 4; j++) {   // B frag: col=mA, j0-7 -> K=kh*8+j, j8-15 -> 16+kh*8+j
            int col = wn * 64 + j * 16 + mA;
            bf[j].q[0] = *(const u32x4*)&Bs[col * 40 + kh * 8];
            bf[j].q[1] = *(const u32x4*)&Bs[col * 40 + 16 + kh * 8];
        }
#pragma unroll
        for (int i = 0; i < 2; i++)
#pragma unroll
            for (int j = 0; j < 4; j++)
                acc[i][j] = wmma_bf16(af[i].v, bf[j].v, acc[i][j]);
    }

    // epilogue: C layout -> element (m = r + 8*kh, n = mA) per accum VGPR r
#pragma unroll
    for (int i = 0; i < 2; i++) {
#pragma unroll
        for (int j = 0; j < 4; j++) {
            int gn = tileN + wn * 64 + j * 16 + mA;
#pragma unroll
            for (int r = 0; r < 8; r++) {
                int gm = tileM + wm * 32 + i * 16 + r + 8 * kh;
                float val = acc[i][j][r];
                int b = gm / NPAD, n = gm % NPAD;
                if (mode == 0) {
                    int t3 = gn / DIM, rem = gn % DIM;
                    int h = rem >> 6, d = rem & 63;
                    unsigned short* dst = (t3 == 0) ? qb : (t3 == 1) ? kb : vb;
                    if (t3 == 0) val *= QSCALE;
                    dst[(((long)b * HEADS + h) * NPAD + n) * HD + d] = f2bf(val);
                } else {
                    if (n < SEQ)
                        out[((long)b * SEQ + n) * DIM + gn] = val + bias[gn];
                }
            }
        }
    }
}

// ---------------------------------------------------------------------------
// Flash attention: one wave per (b, h, 16-query tile). 8 WMMAs / 32-token chunk.
// V chunk staged via async-to-LDS DMA; next chunk's K/V prefetched.
// ---------------------------------------------------------------------------
__global__ __launch_bounds__(32) void attn_kernel(
    const unsigned short* __restrict__ qbuf,
    const unsigned short* __restrict__ kbuf,
    const unsigned short* __restrict__ vbuf,
    const unsigned char*  __restrict__ vmask,   // [B][2101]
    const unsigned char*  __restrict__ tmask,   // [B][20]
    unsigned short* __restrict__ ob)            // [B*NPAD][768] bf16
{
    __shared__ unsigned short Vs[32 * 64];      // V chunk [tok][d]
    __shared__ unsigned short Ps[16 * 48];      // P tile  [m][tok], 96B row stride

    const int lane = threadIdx.x;
    const int tile = blockIdx.x % NTILES;
    const int bh   = blockIdx.x / NTILES;
    const int h    = bh % HEADS;
    const int b    = bh / HEADS;

    // query-tile descriptor: [qs, qs+qv) queries, allowed K = [s1,e1) U [s2,e2)
    int qs, qv, s1, e1, s2, e2;
    if (tile == 0)       { qs = 0;                         qv = 1;  s1 = 0; e1 = SEQ;  s2 = 0;    e2 = 0;    }
    else if (tile < 101) { int i = tile - 1;   qs = 1    + 16 * i; qv = 16; s1 = 1; e1 = 1601; s2 = 2101; e2 = 2121; }
    else if (tile < 126) { int i = tile - 101; qs = 1601 + 16 * i; qv = 16; s1 = 1; e1 = 2001; s2 = 2101; e2 = 2121; }
    else if (tile < 133) { int i = tile - 126; qs = 2001 + 16 * i; qv = (100 - 16 * i) < 16 ? (100 - 16 * i) : 16;
                           s1 = 1; e1 = SEQ; s2 = 0; e2 = 0; }
    else                 { int i = tile - 133; qs = 2101 + 16 * i; qv = (20 - 16 * i) < 16 ? (20 - 16 * i) : 16;
                           s1 = 0; e1 = SEQ; s2 = 0; e2 = 0; }

    const long bhbase = (long)bh * NPAD;
    const unsigned short* qp = qbuf + bhbase * HD;
    const unsigned short* kp = kbuf + bhbase * HD;
    const unsigned short* vp = vbuf + bhbase * HD;
    const unsigned long long Vbase = (unsigned long long)(uintptr_t)vbuf;

    const int n = lane & 15, kh = lane >> 4;

    // Q A-frags for the two 32-wide d-chunks (q pre-scaled by QSCALE)
    FragB qa[2];
    {
        int qrow = qs + n; if (qrow > NPAD - 1) qrow = NPAD - 1;
#pragma unroll
        for (int c = 0; c < 2; c++) {
            const unsigned short* p = qp + (long)qrow * HD + c * 32 + kh * 8;
            qa[c].q[0] = *(const u32x4*)p;
            qa[c].q[1] = *(const u32x4*)(p + 16);
        }
    }

    v8f o[4];
#pragma unroll
    for (int j = 0; j < 4; j++) o[j] = {};
    float mrun[8], lrun[8];
#pragma unroll
    for (int r = 0; r < 8; r++) { mrun[r] = -1e30f; lrun[r] = 0.0f; }

    for (int range = 0; range < 2; range++) {
        int s = range ? s2 : s1, e = range ? e2 : e1;
        if (s >= e) continue;
        for (int t0 = s & ~31; t0 < e; t0 += 32) {
            // stage V chunk [32 tok][64 d] via async DMA: 8 x 16B per lane
            {
                int tok = t0 + lane; if (tok > NPAD - 1) tok = NPAD - 1;
                unsigned ldsv = (unsigned)(uintptr_t)&Vs[lane * 64];
                unsigned voff = (unsigned)((bhbase + tok) * (HD * 2));
                ASYNC_LDS_B128(ldsv, voff, Vbase, 0);
                ASYNC_LDS_B128(ldsv, voff, Vbase, 16);
                ASYNC_LDS_B128(ldsv, voff, Vbase, 32);
                ASYNC_LDS_B128(ldsv, voff, Vbase, 48);
                ASYNC_LDS_B128(ldsv, voff, Vbase, 64);
                ASYNC_LDS_B128(ldsv, voff, Vbase, 80);
                ASYNC_LDS_B128(ldsv, voff, Vbase, 96);
                ASYNC_LDS_B128(ldsv, voff, Vbase, 112);
            }
            // prefetch next chunk's K/V rows into cache (global_prefetch_b8)
            if (t0 + 32 < e) {
                int pt = t0 + 32 + lane; if (pt > NPAD - 1) pt = NPAD - 1;
                __builtin_prefetch(kp + (long)pt * HD, 0, 1);
                __builtin_prefetch(vp + (long)pt * HD, 0, 1);
            }

            // S = Q K^T for the two 16-token tiles of this chunk
            v8f sc0 = {}, sc1 = {};
            int tk0 = t0 + n;      if (tk0 > NPAD - 1) tk0 = NPAD - 1;
            int tk1 = t0 + 16 + n; if (tk1 > NPAD - 1) tk1 = NPAD - 1;
#pragma unroll
            for (int c = 0; c < 2; c++) {
                FragB kf0, kf1;   // B frag: col n = token, K = d
                const unsigned short* p0 = kp + (long)tk0 * HD + c * 32 + kh * 8;
                const unsigned short* p1 = kp + (long)tk1 * HD + c * 32 + kh * 8;
                kf0.q[0] = *(const u32x4*)p0;  kf0.q[1] = *(const u32x4*)(p0 + 16);
                kf1.q[0] = *(const u32x4*)p1;  kf1.q[1] = *(const u32x4*)(p1 + 16);
                sc0 = wmma_bf16(qa[c].v, kf0.v, sc0);
                sc1 = wmma_bf16(qa[c].v, kf1.v, sc1);
            }

            // column masking (range bounds + user masks)
            int col0 = t0 + n, col1 = t0 + 16 + n;
            bool ok0 = (col0 >= s) && (col0 < e);
            bool ok1 = (col1 >= s) && (col1 < e);
            if (ok0) {
                unsigned char mb = (col0 < 2101) ? vmask[b * 2101 + col0]
                                                 : tmask[b * 20 + (col0 - 2101)];
                if (mb) ok0 = false;
            }
            if (ok1) {
                unsigned char mb = (col1 < 2101) ? vmask[b * 2101 + col1]
                                                 : tmask[b * 20 + (col1 - 2101)];
                if (mb) ok1 = false;
            }

            // online softmax (row r lives at vgpr r, lane-half kh -> row r+8*kh)
            float alpha[8];
#pragma unroll
            for (int r = 0; r < 8; r++) {
                float x0 = ok0 ? sc0[r] : -1e30f;
                float x1 = ok1 ? sc1[r] : -1e30f;
                float cm = fmaxf(x0, x1);
#pragma unroll
                for (int off = 1; off < 16; off <<= 1)
                    cm = fmaxf(cm, __shfl_xor(cm, off, 32));
                float mn = fmaxf(mrun[r], cm);
                float al = __expf(mrun[r] - mn);
                float p0 = __expf(x0 - mn);
                float p1 = __expf(x1 - mn);
                float ls = p0 + p1;
#pragma unroll
                for (int off = 1; off < 16; off <<= 1)
                    ls += __shfl_xor(ls, off, 32);
                lrun[r]  = lrun[r] * al + ls;
                mrun[r]  = mn;
                alpha[r] = al;
                int mr = r + 8 * kh;
                Ps[mr * 48 + n]      = f2bf(p0);   // transpose C-layout -> A-layout via LDS
                Ps[mr * 48 + 16 + n] = f2bf(p1);
            }
            WAIT_ASYNC0();
            __syncthreads();

            // P as A-frag (16 q-rows x 32 tokens)
            FragB pf;
            pf.q[0] = *(const u32x4*)&Ps[n * 48 + kh * 16];
            pf.q[1] = *(const u32x4*)&Ps[n * 48 + kh * 16 + 8];

            // rescale O, then O += P * V (4 dim-slabs)
#pragma unroll
            for (int j = 0; j < 4; j++)
#pragma unroll
                for (int r = 0; r < 8; r++) o[j][r] *= alpha[r];

#pragma unroll
            for (int sl = 0; sl < 4; sl++) {
                FragB vf;   // B frag: col n = dim sl*16+n, K = token kh*16+j
#pragma unroll
                for (int j = 0; j < 16; j++)
                    vf.u[j] = Vs[(kh * 16 + j) * 64 + sl * 16 + n];
                o[sl] = wmma_bf16(pf.v, vf.v, o[sl]);
            }
            __syncthreads();
        }
    }

    // write O / l  -> attn_out bf16 [b*NPAD + q][h*64 + d]
#pragma unroll
    for (int sl = 0; sl < 4; sl++) {
#pragma unroll
        for (int r = 0; r < 8; r++) {
            int mr = r + 8 * kh;
            if (mr < qv) {
                float val = o[sl][r] / lrun[r];
                long row = (long)b * NPAD + (qs + mr);
                ob[row * DIM + h * HD + sl * 16 + n] = f2bf(val);
            }
        }
    }
}

// ---------------------------------------------------------------------------
extern "C" void kernel_launch(void* const* d_in, const int* in_sizes, int n_in,
                              void* d_out, int out_size, void* d_ws, size_t ws_size,
                              hipStream_t stream) {
    (void)in_sizes; (void)n_in; (void)out_size; (void)ws_size;
    const float*         src    = (const float*)d_in[0];
    const unsigned char* vmask  = (const unsigned char*)d_in[1];
    const unsigned char* tmask  = (const unsigned char*)d_in[2];
    const float*         w_qkv  = (const float*)d_in[3];
    const float*         w_proj = (const float*)d_in[4];
    const float*         b_proj = (const float*)d_in[5];
    float*               out    = (float*)d_out;

    char* ws = (char*)d_ws;
    size_t off = 0;
    auto alloc = [&](size_t bytes) {
        void* p = ws + off;
        off += (bytes + 255) & ~(size_t)255;
        return p;
    };
    // srcbf doubles as attn_out (src is consumed by the QKV GEMM before attention
    // writes over it; padded rows are re-zeroed by cvt_src every call).
    unsigned short* srcbf = (unsigned short*)alloc((size_t)BATCH * NPAD * DIM * 2);
    unsigned short* wqkv  = (unsigned short*)alloc((size_t)DIM * QKV_N * 2);
    unsigned short* wproj = (unsigned short*)alloc((size_t)DIM * DIM * 2);
    size_t qkvsz = (size_t)BATCH * HEADS * NPAD * HD * 2;
    unsigned short* qbuf = (unsigned short*)alloc(qkvsz);
    unsigned short* kbuf = (unsigned short*)alloc(qkvsz);
    unsigned short* vbuf = (unsigned short*)alloc(qkvsz);

    { int total = BATCH * NPAD * DIM;
      cvt_src_kernel<<<(total + 255) / 256, 256, 0, stream>>>(src, srcbf); }
    { int c = DIM * QKV_N;
      cvt_w_kernel<<<(c + 255) / 256, 256, 0, stream>>>(w_qkv, wqkv, c); }
    { int c = DIM * DIM;
      cvt_w_kernel<<<(c + 255) / 256, 256, 0, stream>>>(w_proj, wproj, c); }

    gemm_bf16_kernel<<<dim3(MROWS / 128, QKV_N / 128), 256, 0, stream>>>(
        srcbf, wqkv, QKV_N, 0, qbuf, kbuf, vbuf, nullptr, nullptr);

    attn_kernel<<<BATCH * HEADS * NTILES, 32, 0, stream>>>(
        qbuf, kbuf, vbuf, vmask, tmask, srcbf);

    gemm_bf16_kernel<<<dim3(MROWS / 128, DIM / 128), 256, 0, stream>>>(
        srcbf, wproj, DIM, 1, nullptr, nullptr, nullptr, b_proj, out);
}
